// GraphKernel_56573309223906
// MI455X (gfx1250) — matrix operand order repaired and checked by reading, not compiled, not run on hardware
//
#include <hip/hip_runtime.h>
#include <hip/hip_bf16.h>

typedef __bf16 bf16_t;
typedef bf16_t v16bf __attribute__((ext_vector_type(16)));
typedef bf16_t v8bf  __attribute__((ext_vector_type(8)));
typedef float  v8f   __attribute__((ext_vector_type(8)));
typedef unsigned int u32x4 __attribute__((ext_vector_type(4)));
typedef int i32x4 __attribute__((ext_vector_type(4)));
typedef int i32x8 __attribute__((ext_vector_type(8)));

#define NODE_DIM 512
#define HIDDEN   4096
#define N_NODES  4096

// TDM availability / flavor detection (ROCm 7.2 = 5-arg builtin; amdgpu-toolchain
// ships the TDM header and uses the 6-arg form).
#if defined(__has_builtin)
#if __has_builtin(__builtin_amdgcn_tensor_load_to_lds)
#define HAVE_TDM 1
#endif
#if __has_builtin(__builtin_amdgcn_s_wait_tensorcnt)
#define HAVE_WAIT_TENSOR 1
#endif
#endif
#if defined(__has_include)
#if __has_include(<hip/amd_detail/amd_gfx1250_TDM.h>)
#define TDM_6ARG 1
#endif
#endif

// LDS A-tile: 64 rows x 512 bf16, rows padded by 16B (TDM pad: 4 DWORDs per 256
// DWORDs) -> stride 1040 B, so rows step 4 banks apart (conflict-free frag reads).
#define LDS_A_STRIDE 1040
#define LDS_A_BYTES  (64 * LDS_A_STRIDE)

// ---------- stage 0a: f32 -> bf16 elementwise (x) ----------
__global__ void k_cvt_bf16(const float* __restrict__ in, bf16_t* __restrict__ out, int n) {
    int i = blockIdx.x * blockDim.x + threadIdx.x;
    if (i < n) out[i] = (bf16_t)in[i];
}

// ---------- stage 0b: transpose (K x N) f32 -> (N x K) bf16 (W1 -> W1^T) ----------
__global__ void k_transpose_bf16(const float* __restrict__ in, bf16_t* __restrict__ out,
                                 int K, int N) {
    __shared__ float tile[32][33];
    int n0 = blockIdx.x * 32;
    int k0 = blockIdx.y * 32;
    int tx = threadIdx.x, ty = threadIdx.y;      // block (32, 8)
    #pragma unroll
    for (int i = 0; i < 4; ++i)
        tile[ty + i * 8][tx] = in[(size_t)(k0 + ty + i * 8) * N + (n0 + tx)];
    __syncthreads();
    #pragma unroll
    for (int i = 0; i < 4; ++i)
        out[(size_t)(n0 + ty + i * 8) * K + (k0 + tx)] = (bf16_t)tile[tx][ty + i * 8];
}

// ---------- stage 1: a = relu(x @ W1 + b1), bf16 WMMA, f32 accumulate ----------
// A  : (4096,512) bf16 row-major    (staged per-block into LDS via TDM)
// Bt : (4096,512) bf16 row-major    (= W1 transposed, so B columns are contiguous)
// C  : (4096,4096) f32
__global__ __launch_bounds__(256)
void k_gemm_relu(const bf16_t* __restrict__ A, const bf16_t* __restrict__ Bt,
                 const float* __restrict__ bias, float* __restrict__ C) {
    const int K = NODE_DIM;
    extern __shared__ unsigned char ldsA[];       // LDS_A_BYTES

    int lane  = threadIdx.x & 31;
    int wv    = threadIdx.x >> 5;
    int lrow  = lane & 15;
    int lhalf = lane >> 4;
    int m0blk = blockIdx.y * 64;
    int m0 = m0blk + (wv & 1) * 32;               // wave tile: 32 rows
    int n0 = blockIdx.x * 256 + (wv >> 1) * 64;   // wave tile: 64 cols

#if HAVE_TDM
    if (wv == 0) {
        // Tensor DMA: 2D tile (512 x 64 rows of bf16) from A[m0blk,0] into LDS,
        // with 16B padding appended per 1024B row.
        unsigned long long ga = (unsigned long long)(const void*)(A + (size_t)m0blk * K);
        unsigned lds_off = (unsigned)(size_t)(void*)ldsA;
        u32x4 g0;
        g0.x = 1u;                                              // count=1, user mode
        g0.y = lds_off;                                         // lds_addr
        g0.z = (unsigned)ga;                                    // global_addr[31:0]
        g0.w = (unsigned)((ga >> 32) & 0x01FFFFFFu) | (2u << 30); // addr[56:32] | type=2
        i32x8 g1;
        g1[0] = (1 << 16)          // data_size = 2 bytes
              | (1 << 20)          // pad_enable
              | (7 << 22)          // pad_interval = 256 DWORDs (one 1024B row)
              | (3 << 25);         // pad_amount   = 4 DWORDs (16B)
        g1[1] = (int)(512u << 16);                  // tensor_dim0[15:0] = 512
        g1[2] = (int)(4096u << 16);                 // dim0 hi=0 | tensor_dim1[15:0]=4096
        g1[3] = (int)(512u << 16);                  // dim1 hi=0 | tile_dim0 = 512
        g1[4] = 64;                                 // tile_dim1 = 64, tile_dim2 = 0
        g1[5] = 512;                                // tensor_dim0_stride = 512 elems
        g1[6] = 0;
        g1[7] = 0;
        i32x4 zz = {0, 0, 0, 0};
#if TDM_6ARG
        i32x8 z8 = {0, 0, 0, 0, 0, 0, 0, 0};
        __builtin_amdgcn_tensor_load_to_lds(g0, g1, zz, zz, z8, 0);
#else
        __builtin_amdgcn_tensor_load_to_lds(g0, g1, zz, zz, 0);
#endif
#if HAVE_WAIT_TENSOR
        __builtin_amdgcn_s_wait_tensorcnt((short)0);
#else
        asm volatile("s_wait_tensorcnt 0x0" ::: "memory");
#endif
    }
    __syncthreads();
#else
    // Cooperative fallback: same padded LDS layout, 16B chunks.
    for (int idx = threadIdx.x; idx < 64 * 64; idx += 256) {
        int r = idx >> 6, c = idx & 63;
        *(v8bf*)(ldsA + r * LDS_A_STRIDE + c * 16) =
            *(const v8bf*)(A + (size_t)(m0blk + r) * K + c * 8);
    }
    __syncthreads();
#endif

    v8f acc[2][4];
    v8f z = {};
    #pragma unroll
    for (int i = 0; i < 2; ++i)
        #pragma unroll
        for (int j = 0; j < 4; ++j) acc[i][j] = z;

    int r0 = (wv & 1) * 32 + lrow;                // local LDS row for this lane
    const unsigned char* arow0 = ldsA + (size_t)r0 * LDS_A_STRIDE;
    const unsigned char* arow1 = ldsA + (size_t)(r0 + 16) * LDS_A_STRIDE;
    const bf16_t* brow[4];
    #pragma unroll
    for (int j = 0; j < 4; ++j)
        brow[j] = Bt + (size_t)(n0 + j * 16 + lrow) * K + lhalf * 16;

    for (int k0 = 0; k0 < K; k0 += 32) {
        // A fragment: lanes 0-15 hold K k0..k0+15, lanes 16-31 hold K k0+16..k0+31
        // (two contiguous 16B chunks, matching the B fragment's K order).
        union { v16bf v; v8bf h[2]; } af[2];
        int kb = (k0 + lhalf * 16) * 2;           // byte offset within LDS row
        af[0].h[0] = *(const v8bf*)(arow0 + kb);
        af[0].h[1] = *(const v8bf*)(arow0 + kb + 16);
        af[1].h[0] = *(const v8bf*)(arow1 + kb);
        af[1].h[1] = *(const v8bf*)(arow1 + kb + 16);
        // B fragment: lane holds column n, 16 contiguous K values (32B load from W1^T).
        v16bf bfr[4];
        #pragma unroll
        for (int j = 0; j < 4; ++j) bfr[j] = *(const v16bf*)(brow[j] + k0);

        if (k0 + 32 < K) {
            __builtin_prefetch(brow[0] + k0 + 32, 0, 3);
            __builtin_prefetch(brow[2] + k0 + 32, 0, 3);
        }

        #pragma unroll
        for (int i = 0; i < 2; ++i)
            #pragma unroll
            for (int j = 0; j < 4; ++j)
                acc[i][j] = __builtin_amdgcn_wmma_f32_16x16x32_bf16(
                    false, af[i].v, false, bfr[j], (short)0, acc[i][j], false, false);
    }

    // C layout: VGPR v -> rows base+v (lanes 0-15) / base+v+8 (lanes 16-31), col = lane&15
    #pragma unroll
    for (int j = 0; j < 4; ++j) {
        int col = n0 + j * 16 + lrow;
        float bz = bias[col];
        #pragma unroll
        for (int i = 0; i < 2; ++i) {
            int rbase = m0 + i * 16 + lhalf * 8;
            #pragma unroll
            for (int v = 0; v < 8; ++v) {
                float val = acc[i][j][v] + bz;
                C[(size_t)(rbase + v) * HIDDEN + col] = val > 0.f ? val : 0.f;
            }
        }
    }
}

// ---------- stage 2: ca[j] = sum_i a[i,j] ----------
__global__ void k_colsum(const float* __restrict__ a, float* __restrict__ ca) {
    int j = blockIdx.x * blockDim.x + threadIdx.x;
    float acc = 0.f;
    for (int i = 0; i < N_NODES; ++i) acc += a[(size_t)i * HIDDEN + j];
    ca[j] = acc;
}

// ---------- column matvec: out[j] = opt_relu(sum_k v[k]*W[k,j] + bscale*b[j]) ----------
__global__ void k_vecmat(const float* __restrict__ v, const float* __restrict__ W,
                         const float* __restrict__ b, float bscale, int do_relu,
                         int K, int N, float* __restrict__ out) {
    int j = blockIdx.x * blockDim.x + threadIdx.x;
    float acc = 0.f;
    for (int k = 0; k < K; ++k) acc += v[k] * W[(size_t)k * N + j];
    acc += bscale * b[j];
    if (do_relu) acc = acc > 0.f ? acc : 0.f;
    out[j] = acc;
}

// ---------- row matvec (wave per row): out[r] = sum_j M[r,j]*vec[j] + addc ----------
__global__ void k_matvec_rows(const float* __restrict__ M, const float* __restrict__ vec,
                              const float* __restrict__ addc, int N,
                              float* __restrict__ out) {
    int lane = threadIdx.x & 31;
    int row  = blockIdx.x * 8 + (threadIdx.x >> 5);
    const float* mrow = M + (size_t)row * N;
    float acc = 0.f;
    for (int j = lane; j < N; j += 32) acc += mrow[j] * vec[j];
    #pragma unroll
    for (int off = 16; off > 0; off >>= 1) acc += __shfl_xor(acc, off, 32);
    if (lane == 0) out[row] = acc + (addc ? addc[0] : 0.f);
}

// ---------- single-block dot: out[0] = u.v + (badd ? badd[0] : 0) ----------
__global__ void k_dot(const float* __restrict__ u, const float* __restrict__ v, int n,
                      const float* __restrict__ badd, float* __restrict__ out) {
    __shared__ float red[256];
    float acc = 0.f;
    for (int i = threadIdx.x; i < n; i += 256) acc += u[i] * v[i];
    red[threadIdx.x] = acc;
    __syncthreads();
    for (int s = 128; s > 0; s >>= 1) {
        if (threadIdx.x < s) red[threadIdx.x] += red[threadIdx.x + s];
        __syncthreads();
    }
    if (threadIdx.x == 0) out[0] = red[0] + (badd ? badd[0] : 0.f);
}

extern "C" void kernel_launch(void* const* d_in, const int* in_sizes, int n_in,
                              void* d_out, int out_size, void* d_ws, size_t ws_size,
                              hipStream_t stream) {
    (void)in_sizes; (void)n_in; (void)out_size; (void)ws_size;
    const float* x  = (const float*)d_in[0];
    // d_in[1] = edge_index (int64) is unused by the reference computation
    const float* W1 = (const float*)d_in[2];
    const float* b1 = (const float*)d_in[3];
    const float* W2 = (const float*)d_in[4];
    const float* b2 = (const float*)d_in[5];
    const float* W3 = (const float*)d_in[6];
    const float* b3 = (const float*)d_in[7];
    const float* W4 = (const float*)d_in[8];
    const float* b4 = (const float*)d_in[9];
    float* out = (float*)d_out;

    char* ws = (char*)d_ws;
    float*  a_buf = (float*)(ws);                              // 64 MB: a = relu(x@W1+b1)
    bf16_t* xb    = (bf16_t*)(ws + 67108864);                  // 4 MB: x in bf16
    bf16_t* w1t   = (bf16_t*)(ws + 67108864 + 4194304);        // 4 MB: W1^T in bf16
    float*  ca    = (float*)(ws + 75497472);                   // colsum(a)
    float*  s     = (float*)(ws + 75497472 + 16384);           // s = ca@W2 + 4096*b2
    float*  t     = (float*)(ws + 75497472 + 32768);           // t = W2@s
    float*  kv    = (float*)(ws + 75497472 + 49152);           // kernel_vector
    float*  p     = (float*)(ws + 75497472 + 65536);           // relu(kv@W3+b3)
    float*  dbs   = (float*)(ws + 75497472 + 81920);           // dot(b2, s)

    k_cvt_bf16<<<(N_NODES * NODE_DIM) / 256, 256, 0, stream>>>(x, xb, N_NODES * NODE_DIM);
    k_transpose_bf16<<<dim3(HIDDEN / 32, NODE_DIM / 32), dim3(32, 8), 0, stream>>>(
        W1, w1t, NODE_DIM, HIDDEN);
    k_gemm_relu<<<dim3(HIDDEN / 256, N_NODES / 64), 256, LDS_A_BYTES, stream>>>(
        xb, w1t, b1, a_buf);
    k_colsum<<<HIDDEN / 256, 256, 0, stream>>>(a_buf, ca);
    // s = colsum(a) @ W2 + N_NODES * b2      (h.sum(0) without materializing h)
    k_vecmat<<<HIDDEN / 256, 256, 0, stream>>>(ca, W2, b2, (float)N_NODES, 0, HIDDEN, HIDDEN, s);
    // t = W2 @ s
    k_matvec_rows<<<HIDDEN / 8, 256, 0, stream>>>(W2, s, nullptr, HIDDEN, t);
    // dot(b2, s) (constant offset of kernel_vector)
    k_dot<<<1, 256, 0, stream>>>(b2, s, HIDDEN, nullptr, dbs);
    // kernel_vector = a @ t + dot(b2, s)
    k_matvec_rows<<<N_NODES / 8, 256, 0, stream>>>(a_buf, t, dbs, HIDDEN, kv);
    // p = relu(kv @ W3 + b3)
    k_vecmat<<<HIDDEN / 256, 256, 0, stream>>>(kv, W3, b3, 1.0f, 1, N_NODES, HIDDEN, p);
    // out = p . W4 + b4
    k_dot<<<1, 256, 0, stream>>>(p, W4, HIDDEN, b4, out);
}
